// GCNSoftmax_43722767073362
// MI455X (gfx1250) — matrix operand order repaired and checked by reading, not compiled, not run on hardware
//
#include <hip/hip_runtime.h>

typedef __bf16 bf16_t;
typedef __attribute__((ext_vector_type(16))) __bf16 v16bf;
typedef __attribute__((ext_vector_type(8)))  float  v8f;

#define IN_FEATS 512
#define HIDDEN   128

// ---------------- init / degrees / norms / CSR ----------------

__global__ __launch_bounds__(256) void zero_int_kernel(int* __restrict__ p, int n) {
    int i = blockIdx.x * blockDim.x + threadIdx.x;
    if (i < n) p[i] = 0;
}

__global__ __launch_bounds__(256) void degree_kernel(const int* __restrict__ src,
                                                     const int* __restrict__ dst,
                                                     int* __restrict__ deg_s,
                                                     int* __restrict__ deg_d, int e) {
    int i = blockIdx.x * blockDim.x + threadIdx.x;
    if (i < e) {
        atomicAdd(&deg_s[src[i]], 1);
        atomicAdd(&deg_d[dst[i]], 1);
    }
}

__global__ __launch_bounds__(256) void norm_kernel(const int* __restrict__ deg_s,
                                                   const int* __restrict__ deg_d,
                                                   float* __restrict__ nsrc,
                                                   float* __restrict__ ndst, int n) {
    int i = blockIdx.x * blockDim.x + threadIdx.x;
    if (i < n) {
        nsrc[i] = rsqrtf(fmaxf((float)deg_s[i], 1.0f));
        ndst[i] = rsqrtf(fmaxf((float)deg_d[i], 1.0f));
    }
}

// single-block exclusive scan of deg_d -> row_start (also primes cursor)
__global__ __launch_bounds__(1024) void scan_kernel(const int* __restrict__ deg,
                                                    int* __restrict__ row_start,
                                                    int* __restrict__ cursor, int n) {
    __shared__ int part[1024];
    int t = threadIdx.x;
    int chunk = (n + 1023) >> 10;
    int b = t * chunk;
    int e_ = b + chunk; if (e_ > n) e_ = n;
    int s = 0;
    for (int i = b; i < e_; ++i) s += deg[i];
    part[t] = s;
    __syncthreads();
    for (int off = 1; off < 1024; off <<= 1) {
        int v = (t >= off) ? part[t - off] : 0;
        __syncthreads();
        part[t] += v;
        __syncthreads();
    }
    int run = (t == 0) ? 0 : part[t - 1];
    for (int i = b; i < e_; ++i) {
        row_start[i] = run;
        cursor[i]    = run;
        run += deg[i];
    }
}

__global__ __launch_bounds__(256) void fill_kernel(const int* __restrict__ src,
                                                   const int* __restrict__ dst,
                                                   int* __restrict__ cursor,
                                                   int* __restrict__ edge_src, int e) {
    int i = blockIdx.x * blockDim.x + threadIdx.x;
    if (i < e) {
        int slot = atomicAdd(&cursor[dst[i]], 1);
        edge_src[slot] = src[i];
    }
}

// ---------------- W1 pre-pack into WMMA B-fragment order ----------------
// flat idx = ((((c*16 + kk)*2 + g)*32 + lane)*8 + e)
//   n = c*16 + (lane&15);  k = kk*32 + 8*(lane>>4) + 16*g + e
__global__ __launch_bounds__(256) void packW1_kernel(const float* __restrict__ W1,
                                                     bf16_t* __restrict__ Bp) {
    int idx  = blockIdx.x * 256 + threadIdx.x;           // 65536 total
    int e    = idx & 7;
    int lane = (idx >> 3) & 31;
    int g    = (idx >> 8) & 1;
    int kk   = (idx >> 9) & 15;
    int c    = (idx >> 13) & 7;
    int ncol = c * 16 + (lane & 15);
    int k    = kk * 32 + ((lane >> 4) << 3) + (g << 4) + e;
    Bp[idx]  = (bf16_t)W1[k * HIDDEN + ncol];
}

// ---------------- GEMM1: XWs = (X @ W1) * norm_src, bf16 WMMA ----------------
// one block = 16 rows x 128 cols; 8 waves, each wave one 16x16 tile; K loop 16x k=32
__global__ __launch_bounds__(256) void gemm1_kernel(const float* __restrict__ X,
                                                    const bf16_t* __restrict__ Bp,
                                                    const float* __restrict__ nsrc,
                                                    float* __restrict__ XWs) {
    __shared__ __align__(16) bf16_t Abuf[16 * IN_FEATS];  // 16 KB
    int t  = threadIdx.x;
    int r0 = blockIdx.x * 16;

    // stage 16x512 f32 tile (contiguous) -> bf16 LDS, coalesced float4 loads
    const float4* Xv = (const float4*)(X + (size_t)r0 * IN_FEATS);
    uint2* Asv = (uint2*)Abuf;
#pragma unroll
    for (int i = 0; i < 8; ++i) {
        int f = i * 256 + t;                 // 2048 float4s
        float4 v = Xv[f];
        union { bf16_t h[4]; uint2 u; } cv;
        cv.h[0] = (bf16_t)v.x; cv.h[1] = (bf16_t)v.y;
        cv.h[2] = (bf16_t)v.z; cv.h[3] = (bf16_t)v.w;
        Asv[f] = cv.u;
    }
    __syncthreads();

    int wave = t >> 5;
    int lane = t & 31;
    int lm   = lane & 15;
    int lh   = lane >> 4;

    v8f acc = {0.f, 0.f, 0.f, 0.f, 0.f, 0.f, 0.f, 0.f};

    for (int kk = 0; kk < 16; ++kk) {
        union { v16bf v; uint4 q[2]; } a, b;
        // A frag: VGPR0-3 -> k = kk*32 + 8*lh + 0..7 ; VGPR4-7 -> +16
        const bf16_t* ab = &Abuf[lm * IN_FEATS + kk * 32 + 8 * lh];
        a.q[0] = *(const uint4*)(ab);
        a.q[1] = *(const uint4*)(ab + 16);
        // B frag: coalesced 16B loads from pre-packed layout
        const bf16_t* bb0 = Bp + ((((size_t)(wave * 16 + kk) * 2 + 0) * 32 + lane) * 8);
        const bf16_t* bb1 = Bp + ((((size_t)(wave * 16 + kk) * 2 + 1) * 32 + lane) * 8);
        b.q[0] = *(const uint4*)bb0;
        b.q[1] = *(const uint4*)bb1;
        acc = __builtin_amdgcn_wmma_f32_16x16x32_bf16(false, a.v, false, b.v,
                                                      (short)0, acc, false, false);
    }

    int n = wave * 16 + lm;
#pragma unroll
    for (int j = 0; j < 8; ++j) {
        int row = r0 + 8 * lh + j;
        XWs[(size_t)row * HIDDEN + n] = acc[j] * nsrc[row];
    }
}

// ---------------- prop1 fused: segment-sum + norm_dst + b1 + ReLU + h@W2 + norm_src ----
// one wave per destination node; lane holds 4 features
__global__ __launch_bounds__(256) void prop1_kernel(const float* __restrict__ XWs,
                                                    const int* __restrict__ row_start,
                                                    const int* __restrict__ deg_d,
                                                    const int* __restrict__ edge_src,
                                                    const float* __restrict__ ndst,
                                                    const float* __restrict__ nsrc,
                                                    const float* __restrict__ b1,
                                                    const float* __restrict__ W2,
                                                    float* __restrict__ ys, int n) {
    int wid  = (blockIdx.x * blockDim.x + threadIdx.x) >> 5;
    int lane = threadIdx.x & 31;
    if (wid >= n) return;
    int beg = row_start[wid];
    int cnt = deg_d[wid];
    int f0  = lane * 4;
    float a0 = 0.f, a1 = 0.f, a2 = 0.f, a3 = 0.f;
    for (int i = 0; i < cnt; ++i) {
        int s = edge_src[beg + i];
        float4 v = *(const float4*)(XWs + (size_t)s * HIDDEN + f0);
        a0 += v.x; a1 += v.y; a2 += v.z; a3 += v.w;
    }
    float nd = ndst[wid];
    float4 bb = *(const float4*)(b1 + f0);
    float h0 = fmaxf(a0 * nd + bb.x, 0.f);
    float h1 = fmaxf(a1 * nd + bb.y, 0.f);
    float h2 = fmaxf(a2 * nd + bb.z, 0.f);
    float h3 = fmaxf(a3 * nd + bb.w, 0.f);
    const float2* W2v = (const float2*)W2;     // [128][2]
    float2 w0 = W2v[f0 + 0], w1 = W2v[f0 + 1], w2 = W2v[f0 + 2], w3 = W2v[f0 + 3];
    float p0 = h0 * w0.x + h1 * w1.x + h2 * w2.x + h3 * w3.x;
    float p1 = h0 * w0.y + h1 * w1.y + h2 * w2.y + h3 * w3.y;
#pragma unroll
    for (int off = 16; off > 0; off >>= 1) {
        p0 += __shfl_xor(p0, off, 32);
        p1 += __shfl_xor(p1, off, 32);
    }
    if (lane == 0) {
        float s = nsrc[wid];
        ys[wid * 2 + 0] = p0 * s;
        ys[wid * 2 + 1] = p1 * s;
    }
}

// ---------------- prop2 fused: segment-sum + norm_dst + b2 + softmax ----------------
__global__ __launch_bounds__(256) void prop2_kernel(const float* __restrict__ ys,
                                                    const int* __restrict__ row_start,
                                                    const int* __restrict__ deg_d,
                                                    const int* __restrict__ edge_src,
                                                    const float* __restrict__ ndst,
                                                    const float* __restrict__ b2,
                                                    float* __restrict__ out, int n) {
    int i = blockIdx.x * blockDim.x + threadIdx.x;
    if (i >= n) return;
    int beg = row_start[i];
    int cnt = deg_d[i];
    float a0 = 0.f, a1 = 0.f;
    const float2* yv = (const float2*)ys;
    for (int j = 0; j < cnt; ++j) {
        float2 v = yv[edge_src[beg + j]];
        a0 += v.x; a1 += v.y;
    }
    float nd = ndst[i];
    float l0 = a0 * nd + b2[0];
    float l1 = a1 * nd + b2[1];
    float m  = fmaxf(l0, l1);
    float e0 = __expf(l0 - m);
    float e1 = __expf(l1 - m);
    float inv = 1.0f / (e0 + e1);
    out[2 * i + 0] = e0 * inv;
    out[2 * i + 1] = e1 * inv;
}

// ---------------- launcher ----------------

extern "C" void kernel_launch(void* const* d_in, const int* in_sizes, int n_in,
                              void* d_out, int out_size, void* d_ws, size_t ws_size,
                              hipStream_t stream) {
    const float* X  = (const float*)d_in[0];
    const float* W1 = (const float*)d_in[1];
    const float* b1 = (const float*)d_in[2];
    const float* W2 = (const float*)d_in[3];
    const float* b2 = (const float*)d_in[4];
    const int*   src = (const int*)d_in[5];
    const int*   dst = (const int*)d_in[6];

    const int N = in_sizes[0] / IN_FEATS;   // 100000 (multiple of 16)
    const int E = in_sizes[5];              // 1600000

    char* p = (char*)d_ws;
    int*    deg_s    = (int*)p;   p += (size_t)N * 4;
    int*    deg_d    = (int*)p;   p += (size_t)N * 4;   // contiguous with deg_s
    float*  nsrc     = (float*)p; p += (size_t)N * 4;
    float*  ndst     = (float*)p; p += (size_t)N * 4;
    int*    row_start= (int*)p;   p += (size_t)N * 4;
    int*    cursor   = (int*)p;   p += (size_t)N * 4;
    int*    edge_src = (int*)p;   p += (size_t)E * 4;
    bf16_t* Bp       = (bf16_t*)p; p += (size_t)IN_FEATS * HIDDEN * 2;
    float*  XWs      = (float*)p; p += (size_t)N * HIDDEN * 4;
    float*  ys       = (float*)p; p += (size_t)N * 2 * 4;

    const int tb = 256;
    zero_int_kernel<<<(2 * N + tb - 1) / tb, tb, 0, stream>>>(deg_s, 2 * N);
    degree_kernel<<<(E + tb - 1) / tb, tb, 0, stream>>>(src, dst, deg_s, deg_d, E);
    norm_kernel<<<(N + tb - 1) / tb, tb, 0, stream>>>(deg_s, deg_d, nsrc, ndst, N);
    scan_kernel<<<1, 1024, 0, stream>>>(deg_d, row_start, cursor, N);
    fill_kernel<<<(E + tb - 1) / tb, tb, 0, stream>>>(src, dst, cursor, edge_src, E);
    packW1_kernel<<<(IN_FEATS * HIDDEN) / tb, tb, 0, stream>>>(W1, Bp);
    gemm1_kernel<<<N / 16, tb, 0, stream>>>(X, Bp, nsrc, XWs);
    prop1_kernel<<<((size_t)N * 32 + tb - 1) / tb, tb, 0, stream>>>(
        XWs, row_start, deg_d, edge_src, ndst, nsrc, b1, W2, ys, N);
    prop2_kernel<<<(N + tb - 1) / tb, tb, 0, stream>>>(
        ys, row_start, deg_d, edge_src, ndst, b2, (float*)d_out, N);
}
